// DecoderBlock_41446434406617
// MI455X (gfx1250) — matrix operand order repaired and checked by reading, not compiled, not run on hardware
//
#include <hip/hip_runtime.h>

// ---------------------------------------------------------------------------
// MI455X (gfx1250) decoder block: TDM-staged bf16 WMMA GEMMs + flash attention
// ---------------------------------------------------------------------------

typedef __attribute__((ext_vector_type(16))) __bf16 v16bf;
typedef __attribute__((ext_vector_type(8)))  __bf16 v8bf;
typedef __attribute__((ext_vector_type(4)))  __bf16 v4bf;
typedef __attribute__((ext_vector_type(8)))  float  v8f;
typedef __attribute__((ext_vector_type(4)))  float  v4f;
typedef __attribute__((ext_vector_type(4)))  unsigned int v4u;
typedef __attribute__((ext_vector_type(8)))  int    v8i;
typedef __attribute__((ext_vector_type(4)))  int    v4i;

constexpr int Bc  = 4;
constexpr int Sc  = 2048;
constexpr int Dc  = 2048;
constexpr int Hc  = 16;
constexpr int DHc = 128;
constexpr int FFc = 8192;
constexpr int NT  = Bc * Sc;          // 8192 tokens
constexpr float EPSc = 1e-5f;

__device__ __forceinline__ v8f wmma_bf16(v16bf a, v16bf b, v8f c) {
  return __builtin_amdgcn_wmma_f32_16x16x32_bf16(
      false, a, false, b, (short)0, c, false, false);
}

// Assemble a v16bf A-fragment from two 16-byte aligned groups of 8.
__device__ __forceinline__ v16bf ldA2(const __bf16* p0, const __bf16* p1) {
  union { v16bf v; v8bf h[2]; } u;
  u.h[0] = *(const v8bf*)p0;
  u.h[1] = *(const v8bf*)p1;
  return u.v;
}

// ---------------------------------------------------------------------------
// Tensor Data Mover: 2D tile (tile_w x tile_h, bf16) global -> LDS.
// Hand-packed D# per CDNA5 ISA ch.8 (group0 + group1, groups 2/3 zero => 2D).
// ---------------------------------------------------------------------------
__device__ __forceinline__ void tdm_load_2d_bf16(
    unsigned lds_addr, const void* gptr,
    unsigned tile_w, unsigned tile_h,
    unsigned tensor_w, unsigned tensor_h,
    unsigned long long row_stride_elems)
{
  unsigned long long ga = (unsigned long long)(uintptr_t)gptr;
  v4u g0;
  g0[0] = 1u;                                   // count=1 (valid descriptor)
  g0[1] = lds_addr;                             // [63:32]  lds_addr (bytes)
  g0[2] = (unsigned)(ga & 0xFFFFFFFFull);       // [95:64]  global_addr lo
  g0[3] = (unsigned)((ga >> 32) & 0x01FFFFFFull) | 0x80000000u; // addr hi + type=2

  v8i g1;
  g1[0] = (int)(1u << 16);                      // data_size=1 (2 bytes/elem)
  g1[1] = (int)((tensor_w & 0xFFFFu) << 16);    // tensor_dim0[15:0] @ [79:64]... bits 79:48
  g1[2] = (int)(((tensor_w >> 16) & 0xFFFFu) |
                ((tensor_h & 0xFFFFu) << 16));  // tensor_dim0 hi | tensor_dim1 lo
  g1[3] = (int)(((tensor_h >> 16) & 0xFFFFu) |
                ((tile_w & 0xFFFFu) << 16));    // tensor_dim1 hi | tile_dim0
  g1[4] = (int)(tile_h & 0xFFFFu);              // tile_dim1 (tile_dim2 = 0)
  g1[5] = (int)(unsigned)(row_stride_elems & 0xFFFFFFFFull);      // dim0_stride lo
  g1[6] = (int)(unsigned)((row_stride_elems >> 32) & 0xFFFFull);  // dim0_stride hi
  g1[7] = 0;

  v4i z4 = {0, 0, 0, 0};
#if defined(__clang_major__) && (__clang_major__ >= 23)
  v8i z8 = {0, 0, 0, 0, 0, 0, 0, 0};
  __builtin_amdgcn_tensor_load_to_lds(g0, g1, z4, z4, z8, 0);
#else
  __builtin_amdgcn_tensor_load_to_lds(g0, g1, z4, z4, 0);
#endif
}

// ---------------------------------------------------------------------------
// fp32 -> bf16 conversion (vectorized x4)
// ---------------------------------------------------------------------------
__global__ void cvt_f32_bf16(const v4f* __restrict__ in, v4bf* __restrict__ out, long n4) {
  long i  = (long)blockIdx.x * blockDim.x + threadIdx.x;
  long st = (long)gridDim.x * blockDim.x;
  for (; i < n4; i += st) {
    v4f x = in[i];
    v4bf o = { (__bf16)x[0], (__bf16)x[1], (__bf16)x[2], (__bf16)x[3] };
    out[i] = o;
  }
}

// ---------------------------------------------------------------------------
// GEMM: C[M,N] = A[M,K] * W[N,K]^T + bias
// Block tile 64(M) x 128(N); K staged in 32-wide chunks via TDM into LDS,
// double buffered. 8 waves = 4(M) x 2(N); each wave owns a 16x64 strip.
// EPI: 0 = fp32 out, 2 = ReLU -> bf16 out, 3 = bf16 out transposed [B,H,DH,S]
// ---------------------------------------------------------------------------
template <int EPI>
__global__ __launch_bounds__(256) void gemm_bf16_wmma(
    const __bf16* __restrict__ A, const __bf16* __restrict__ W,
    const float* __restrict__ bias, float* __restrict__ Cf,
    __bf16* __restrict__ Cb, int M, int N, int K)
{
  extern __shared__ char smem[];                // 2 * (4KB A + 8KB W) = 24KB
  constexpr int STG = 64*32*2 + 128*32*2;       // 12288 bytes per stage

  const int lane = threadIdx.x & 31;
  const int wid  = threadIdx.x >> 5;
  const int lo   = lane & 15;
  const int hi   = lane >> 4;
  const int mw   = wid & 3;                     // wave M-tile (0..3)
  const int nw   = wid >> 2;                    // wave N-strip (0..1)

  const int nbn  = N >> 7;
  const int bm   = blockIdx.x / nbn;
  const int bn   = blockIdx.x % nbn;
  const int row0 = bm * 64;
  const int col0 = bn * 128;

  const unsigned ldsbase = (unsigned)(uintptr_t)(void*)smem;  // flat->LDS offset
  const __bf16* Ablk = A + (size_t)row0 * K;
  const __bf16* Wblk = W + (size_t)col0 * K;

  v8f z = {0.f,0.f,0.f,0.f,0.f,0.f,0.f,0.f};
  v8f acc[4];
#pragma unroll
  for (int t = 0; t < 4; ++t) acc[t] = z;

  const int nk = K >> 5;
  if (wid == 0) {
    tdm_load_2d_bf16(ldsbase,        Ablk, 32, 64,  (unsigned)K, (unsigned)M, (unsigned long long)K);
    tdm_load_2d_bf16(ldsbase + 4096, Wblk, 32, 128, (unsigned)K, (unsigned)N, (unsigned long long)K);
  }

  for (int kI = 0; kI < nk; ++kI) {
    const int cur = kI & 1;
    if (wid == 0) __builtin_amdgcn_s_wait_tensorcnt(0);
    __syncthreads();                            // stage `cur` ready; `1-cur` free
    if (wid == 0 && (kI + 1) < nk) {
      const int k1 = (kI + 1) << 5;
      tdm_load_2d_bf16(ldsbase + (1-cur)*STG,        Ablk + k1, 32, 64,  (unsigned)K, (unsigned)M, (unsigned long long)K);
      tdm_load_2d_bf16(ldsbase + (1-cur)*STG + 4096, Wblk + k1, 32, 128, (unsigned)K, (unsigned)N, (unsigned long long)K);
    }

    const __bf16* sA = (const __bf16*)(smem + cur * STG);
    const __bf16* sW = (const __bf16*)(smem + cur * STG + 4096);
    // A fragment (16x32) for this wave's M-tile: 16 contiguous K per lane
    const __bf16* ap = sA + (mw*16 + lo) * 32 + hi*16;
    v16bf a = ldA2(ap, ap + 8);
#pragma unroll
    for (int t = 0; t < 4; ++t) {
      // B fragment (32x16): lane lo = col, 16 contiguous K per lane
      v16bf b = *(const v16bf*)(sW + (nw*64 + t*16 + lo) * 32 + hi*16);
      acc[t] = wmma_bf16(a, b, acc[t]);
    }
  }

#pragma unroll
  for (int t = 0; t < 4; ++t) {
    const int col = col0 + nw*64 + t*16 + lo;
    const float bias_v = bias ? bias[col] : 0.0f;
#pragma unroll
    for (int r = 0; r < 8; ++r) {
      const int row = row0 + mw*16 + r + hi*8;  // C layout: VGPR r -> M=r / M=r+8
      float y = acc[t][r] + bias_v;
      if (EPI == 0) {
        Cf[(size_t)row * N + col] = y;
      } else if (EPI == 2) {
        y = fmaxf(y, 0.0f);
        Cb[(size_t)row * N + col] = (__bf16)y;
      } else if (EPI == 3) {
        const int bi = row >> 11;               // row / Sc
        const int s  = row & (Sc - 1);
        const int hh = col >> 7;                // col / DHc
        const int dh = col & (DHc - 1);
        Cb[(((size_t)(bi * Hc + hh)) * DHc + dh) * Sc + s] = (__bf16)y;
      }
    }
  }
}

// ---------------------------------------------------------------------------
// Per-(token, head) LayerNorm over DH=128, one wave32 per head-vector.
// ---------------------------------------------------------------------------
__global__ __launch_bounds__(256) void qk_ln(
    const float* __restrict__ in, __bf16* __restrict__ out, float scale, int total)
{
  const int wid  = threadIdx.x >> 5;
  const int lane = threadIdx.x & 31;
  const long gw  = (long)blockIdx.x * 8 + wid;
  if (gw >= total) return;

  v4f x = *(const v4f*)(in + gw * DHc + lane * 4);
  float s = x[0] + x[1] + x[2] + x[3];
#pragma unroll
  for (int m = 1; m < 32; m <<= 1) s += __shfl_xor(s, m);
  const float mu = s * (1.0f / DHc);

  const float d0 = x[0]-mu, d1 = x[1]-mu, d2 = x[2]-mu, d3 = x[3]-mu;
  float vs = d0*d0 + d1*d1 + d2*d2 + d3*d3;
#pragma unroll
  for (int m = 1; m < 32; m <<= 1) vs += __shfl_xor(vs, m);
  const float inv = rsqrtf(vs * (1.0f / DHc) + EPSc) * scale;

  v4bf o = { (__bf16)(d0*inv), (__bf16)(d1*inv), (__bf16)(d2*inv), (__bf16)(d3*inv) };
  *(v4bf*)(out + gw * DHc + lane * 4) = o;
}

// ---------------------------------------------------------------------------
// Residual + LayerNorm over D=2048, one 256-thread block per token.
// ---------------------------------------------------------------------------
__global__ __launch_bounds__(256) void add_ln(
    const float* __restrict__ a, const float* __restrict__ b,
    float* __restrict__ outf, __bf16* __restrict__ outb)
{
  __shared__ float redA[8];
  __shared__ float redB[8];
  const long t   = blockIdx.x;
  const int tid  = threadIdx.x;
  const int wid  = tid >> 5;
  const int lane = tid & 31;

  const float* pa = a + t * Dc;
  const float* pb = b + t * Dc;
  float x[8];
  float s = 0.f;
#pragma unroll
  for (int i = 0; i < 8; ++i) { x[i] = pa[tid + i*256] + pb[tid + i*256]; s += x[i]; }
#pragma unroll
  for (int m = 1; m < 32; m <<= 1) s += __shfl_xor(s, m);
  if (lane == 0) redA[wid] = s;
  __syncthreads();
  s = redA[0]+redA[1]+redA[2]+redA[3]+redA[4]+redA[5]+redA[6]+redA[7];
  const float mu = s * (1.0f / Dc);

  float vs = 0.f;
#pragma unroll
  for (int i = 0; i < 8; ++i) { const float d = x[i]-mu; vs += d*d; }
#pragma unroll
  for (int m = 1; m < 32; m <<= 1) vs += __shfl_xor(vs, m);
  if (lane == 0) redB[wid] = vs;
  __syncthreads();
  vs = redB[0]+redB[1]+redB[2]+redB[3]+redB[4]+redB[5]+redB[6]+redB[7];
  const float inv = rsqrtf(vs * (1.0f / Dc) + EPSc);

#pragma unroll
  for (int i = 0; i < 8; ++i) {
    const float y = (x[i]-mu) * inv;
    outf[t * Dc + tid + i*256] = y;
    if (outb) outb[t * Dc + tid + i*256] = (__bf16)y;
  }
}

// ---------------------------------------------------------------------------
// Flash attention: one wave per (batch, head, 16-query tile).
// qb/kb: [B*S, D] bf16 (LayerNormed, q pre-scaled by 1/sqrt(DH))
// vtb  : [B, H, DH, S] bf16 (transposed by V-GEMM epilogue)
// ---------------------------------------------------------------------------
__global__ __launch_bounds__(256) void attn_wmma(
    const __bf16* __restrict__ qb, const __bf16* __restrict__ kb,
    const __bf16* __restrict__ vtb, __bf16* __restrict__ ctxb, int total)
{
  __shared__ __bf16 ldsP[8][16 * 32];           // per-wave P tile (16 q x 32 keys)

  const int wid  = threadIdx.x >> 5;
  const int lane = threadIdx.x & 31;
  const int lo   = lane & 15;
  const int hi   = lane >> 4;
  const int gw   = blockIdx.x * 8 + wid;
  if (gw >= total) return;

  const int qt = gw & 127;
  const int h  = (gw >> 7) & (Hc - 1);
  const int bi = gw >> 11;

  v16bf aq[4];
  {
    const __bf16* qrow = qb + ((size_t)(bi * Sc + qt * 16 + lo) * Dc + h * DHc);
#pragma unroll
    for (int c = 0; c < 4; ++c)
      aq[c] = ldA2(qrow + c*32 + hi*16, qrow + c*32 + hi*16 + 8);
  }

  v8f z = {0.f,0.f,0.f,0.f,0.f,0.f,0.f,0.f};
  v8f acc[8];
#pragma unroll
  for (int t = 0; t < 8; ++t) acc[t] = z;
  float rmax[8], rsum[8];
#pragma unroll
  for (int r = 0; r < 8; ++r) { rmax[r] = -1e30f; rsum[r] = 0.f; }

  const int nkb = (qt + 2) >> 1;
  for (int kbI = 0; kbI < nkb; ++kbI) {
    const int kbase = kbI * 32;

    v8f s0 = z, s1 = z;
#pragma unroll
    for (int c = 0; c < 4; ++c) {
      const __bf16* kr0 = kb + ((size_t)(bi * Sc + kbase + lo)      * Dc + h*DHc + c*32 + hi*16);
      const __bf16* kr1 = kb + ((size_t)(bi * Sc + kbase + 16 + lo) * Dc + h*DHc + c*32 + hi*16);
      s0 = wmma_bf16(aq[c], *(const v16bf*)kr0, s0);
      s1 = wmma_bf16(aq[c], *(const v16bf*)kr1, s1);
    }

    float mx[8];
#pragma unroll
    for (int r = 0; r < 8; ++r) {
      const int qi = qt * 16 + r + hi * 8;
      const float a0 = (kbase + lo      <= qi) ? s0[r] : -1e30f;
      const float a1 = (kbase + 16 + lo <= qi) ? s1[r] : -1e30f;
      s0[r] = a0; s1[r] = a1;
      mx[r] = fmaxf(a0, a1);
    }
#pragma unroll
    for (int m = 1; m < 16; m <<= 1)
#pragma unroll
      for (int r = 0; r < 8; ++r) mx[r] = fmaxf(mx[r], __shfl_xor(mx[r], m));

    float alpha[8], psum[8];
#pragma unroll
    for (int r = 0; r < 8; ++r) {
      const float mn = fmaxf(rmax[r], mx[r]);
      alpha[r] = __expf(rmax[r] - mn);
      rmax[r]  = mn;
      const float p0 = __expf(s0[r] - mn);
      const float p1 = __expf(s1[r] - mn);
      s0[r] = p0; s1[r] = p1;
      psum[r] = p0 + p1;
    }
#pragma unroll
    for (int m = 1; m < 16; m <<= 1)
#pragma unroll
      for (int r = 0; r < 8; ++r) psum[r] += __shfl_xor(psum[r], m);
#pragma unroll
    for (int r = 0; r < 8; ++r) rsum[r] = rsum[r] * alpha[r] + psum[r];
#pragma unroll
    for (int t = 0; t < 8; ++t)
#pragma unroll
      for (int r = 0; r < 8; ++r) acc[t][r] *= alpha[r];

    __bf16* Pw = &ldsP[wid][0];
#pragma unroll
    for (int r = 0; r < 8; ++r) {
      const int row = r + hi * 8;
      Pw[row * 32 + lo]      = (__bf16)s0[r];
      Pw[row * 32 + 16 + lo] = (__bf16)s1[r];
    }
    __builtin_amdgcn_wave_barrier();
    v16bf pa = ldA2(Pw + lo*32 + hi*16, Pw + lo*32 + hi*16 + 8);
    __builtin_amdgcn_wave_barrier();

#pragma unroll
    for (int t = 0; t < 8; ++t) {
      const __bf16* vr = vtb +
          (((size_t)(bi * Hc + h) * DHc + t*16 + lo) * Sc + kbase + hi*16);
      acc[t] = wmma_bf16(pa, *(const v16bf*)vr, acc[t]);
    }
  }

#pragma unroll
  for (int t = 0; t < 8; ++t) {
#pragma unroll
    for (int r = 0; r < 8; ++r) {
      const int row = qt * 16 + r + hi * 8;
      const int col = h * DHc + t * 16 + lo;
      const float y = acc[t][r] / rsum[r];
      ctxb[(size_t)(bi * Sc + row) * Dc + col] = (__bf16)y;
    }
  }
}

// ---------------------------------------------------------------------------
// Host orchestration
// ---------------------------------------------------------------------------
extern "C" void kernel_launch(void* const* d_in, const int* in_sizes, int n_in,
                              void* d_out, int out_size, void* d_ws, size_t ws_size,
                              hipStream_t stream)
{
  (void)in_sizes; (void)n_in; (void)out_size; (void)ws_size;
  const float* x  = (const float*)d_in[0];
  const float* Wq = (const float*)d_in[2];  const float* bq = (const float*)d_in[3];
  const float* Wk = (const float*)d_in[4];  const float* bk = (const float*)d_in[5];
  const float* Wv = (const float*)d_in[6];  const float* bv = (const float*)d_in[7];
  const float* Wo = (const float*)d_in[8];  const float* bo = (const float*)d_in[9];
  const float* W1 = (const float*)d_in[10]; const float* b1 = (const float*)d_in[11];
  const float* W2 = (const float*)d_in[12]; const float* b2 = (const float*)d_in[13];
  float* out = (float*)d_out;

  char* ws = (char*)d_ws;
  size_t off = 0;
  auto alloc = [&](size_t bytes) -> char* {
    char* p = ws + off;
    off += (bytes + 255) & ~(size_t)255;
    return p;
  };

  __bf16* xb   = (__bf16*)alloc((size_t)NT * Dc * 2);
  __bf16* wqb  = (__bf16*)alloc((size_t)Dc * Dc * 2);
  __bf16* wkb  = (__bf16*)alloc((size_t)Dc * Dc * 2);
  __bf16* wvb  = (__bf16*)alloc((size_t)Dc * Dc * 2);
  __bf16* wob  = (__bf16*)alloc((size_t)Dc * Dc * 2);
  __bf16* w1b  = (__bf16*)alloc((size_t)FFc * Dc * 2);
  __bf16* w2b  = (__bf16*)alloc((size_t)Dc * FFc * 2);
  float*  q32  = (float*)alloc((size_t)NT * Dc * 4);
  float*  k32  = (float*)alloc((size_t)NT * Dc * 4);
  __bf16* qb   = (__bf16*)alloc((size_t)NT * Dc * 2);
  __bf16* kbf  = (__bf16*)alloc((size_t)NT * Dc * 2);
  __bf16* vtb  = (__bf16*)alloc((size_t)NT * Dc * 2);
  __bf16* ctxb = (__bf16*)alloc((size_t)NT * Dc * 2);
  float*  x1   = (float*)alloc((size_t)NT * Dc * 4);
  __bf16* hb   = (__bf16*)alloc((size_t)NT * FFc * 2);
  float*  attn32 = q32;     // q32 dead after qk_ln
  float*  ff32   = k32;     // k32 dead after qk_ln
  __bf16* x1b    = qb;      // qb dead after attention

  cvt_f32_bf16<<<2048, 256, 0, stream>>>((const v4f*)x,  (v4bf*)xb,  (long)NT * Dc / 4);
  cvt_f32_bf16<<<1024, 256, 0, stream>>>((const v4f*)Wq, (v4bf*)wqb, (long)Dc * Dc / 4);
  cvt_f32_bf16<<<1024, 256, 0, stream>>>((const v4f*)Wk, (v4bf*)wkb, (long)Dc * Dc / 4);
  cvt_f32_bf16<<<1024, 256, 0, stream>>>((const v4f*)Wv, (v4bf*)wvb, (long)Dc * Dc / 4);
  cvt_f32_bf16<<<1024, 256, 0, stream>>>((const v4f*)Wo, (v4bf*)wob, (long)Dc * Dc / 4);
  cvt_f32_bf16<<<2048, 256, 0, stream>>>((const v4f*)W1, (v4bf*)w1b, (long)FFc * Dc / 4);
  cvt_f32_bf16<<<2048, 256, 0, stream>>>((const v4f*)W2, (v4bf*)w2b, (long)Dc * FFc / 4);

  // blocks = (M/64) * (N/128); 24KB dynamic LDS for double-buffered TDM stages
  const int gD  = (NT / 64) * (Dc / 128);     // 2048
  const int gFF = (NT / 64) * (FFc / 128);    // 8192
  const size_t SMEM = 2 * (64*32*2 + 128*32*2);

  gemm_bf16_wmma<0><<<gD, 256, SMEM, stream>>>(xb, wqb, bq, q32, nullptr, NT, Dc, Dc);
  gemm_bf16_wmma<0><<<gD, 256, SMEM, stream>>>(xb, wkb, bk, k32, nullptr, NT, Dc, Dc);
  gemm_bf16_wmma<3><<<gD, 256, SMEM, stream>>>(xb, wvb, bv, nullptr, vtb, NT, Dc, Dc);

  const int lnWaves = NT * Hc;
  qk_ln<<<lnWaves / 8, 256, 0, stream>>>(q32, qb, 0.08838834764831845f, lnWaves);
  qk_ln<<<lnWaves / 8, 256, 0, stream>>>(k32, kbf, 1.0f, lnWaves);

  const int attnWaves = Bc * Hc * (Sc / 16);
  attn_wmma<<<attnWaves / 8, 256, 0, stream>>>(qb, kbf, vtb, ctxb, attnWaves);

  gemm_bf16_wmma<0><<<gD, 256, SMEM, stream>>>(ctxb, wob, bo, attn32, nullptr, NT, Dc, Dc);
  add_ln<<<NT, 256, 0, stream>>>(x, attn32, x1, x1b);

  gemm_bf16_wmma<2><<<gFF, 256, SMEM, stream>>>(x1b, w1b, b1, nullptr, hb, NT, FFc, Dc);
  gemm_bf16_wmma<0><<<gD,  256, SMEM, stream>>>(hb, w2b, b2, ff32, nullptr, NT, Dc, FFc);
  add_ln<<<NT, 256, 0, stream>>>(x1, ff32, out, nullptr);
}
